// PairwiseAdjDecoder_51668456571285
// MI455X (gfx1250) — matrix operand order repaired and checked by reading, not compiled, not run on hardware
//
#include <hip/hip_runtime.h>
#include <hip/hip_bf16.h>

#define Bsz 512
#define Dsz 256
#define Hsz 512

// ---- WMMA vector types (gfx1250, wave32) ----
typedef __attribute__((ext_vector_type(16))) __bf16 v16bf;
typedef __attribute__((ext_vector_type(8)))  float  v8f;

union FragBF { uint4 u[2]; v16bf v; };

__device__ __forceinline__ unsigned short f32_to_bf16_rne(float f) {
    union { float f; unsigned u; } v; v.f = f;
    unsigned r = v.u + 0x7FFFu + ((v.u >> 16) & 1u);
    return (unsigned short)(r >> 16);
}

// Branchless tanh on the hardware TRANS pipe:
//   tanh(x) = sign(x) * (t-1)/(t+1),  t = 2^(min(|x|*2*log2(e), 126))
// One v_exp_f32 + one v_rcp_f32, ~8 VALU, no exec-mask branches.
__device__ __forceinline__ float fast_tanhf(float x) {
    float ax = __builtin_fabsf(x);
    float a  = __builtin_fminf(ax * 2.885390081777927f, 126.0f); // 2*log2(e)*|x|
    float t  = __builtin_amdgcn_exp2f(a);                        // v_exp_f32
    float r  = (t - 1.0f) * __builtin_amdgcn_rcpf(t + 1.0f);     // v_rcp_f32
    return __builtin_copysignf(r, x);
}

// ------------------------------------------------------------------
// Prep 1: Yi[i,k] = h[i,:] @ W1[:D,k] + b1[k]   (f32)
//         Yj[j,k] = h[j,:] @ W1[D:,k]           (f32)
// ------------------------------------------------------------------
__global__ __launch_bounds__(256)
void prep_y_kernel(const float* __restrict__ h, const float* __restrict__ W1,
                   const float* __restrict__ b1,
                   float* __restrict__ Yi, float* __restrict__ Yj) {
    int idx = blockIdx.x * 256 + threadIdx.x;           // 0 .. 2*B*H-1
    int which = idx >> 18;                              // 0 -> Yi, 1 -> Yj
    int e = idx & (Bsz * Hsz - 1);
    int row = e >> 9;
    int k   = e & (Hsz - 1);
    const float* hr = h + row * Dsz;
    const float* w  = W1 + (which ? Dsz * Hsz : 0) + k; // W1[d][k], stride H
    float s = which ? 0.0f : b1[k];
    #pragma unroll 8
    for (int d = 0; d < Dsz; ++d) s = fmaf(hr[d], w[d * Hsz], s);
    (which ? Yj : Yi)[e] = s;
}

// ------------------------------------------------------------------
// Prep 2: W2T[n][k] = bf16(W2[k][n])  (transposed, contiguous in K)
// ------------------------------------------------------------------
__global__ __launch_bounds__(256)
void w2t_kernel(const float* __restrict__ W2, unsigned short* __restrict__ W2T) {
    int idx = blockIdx.x * 256 + threadIdx.x;           // 0 .. H*H-1
    int n = idx >> 9;
    int k = idx & (Hsz - 1);
    W2T[n * Hsz + k] = f32_to_bf16_rne(W2[k * Hsz + n]);
}

// ------------------------------------------------------------------
// Main fused kernel.
// Block = 8 wave32s = 4 M-waves x 2 N-waves over a 16i x 8j = 128-row
// pair block. Each wave: 2 M-subtiles (32 rows) x 2 N-chunks per pass,
// so each fragment feeds two wmmas:
//   per k-step: 4 ds_load_b128 (A) + 4 global_load_b128 (B) -> 4 wmma
// Column-partial row sums combined across N-waves via ds_add_f32.
// ------------------------------------------------------------------
#define LDS_STRIDE 520   // byte stride 1040, 16B aligned

__global__ __launch_bounds__(256)
void pair_mlp_kernel(const float* __restrict__ Yi, const float* __restrict__ Yj,
                     const unsigned short* __restrict__ W2T,
                     const float* __restrict__ b2, const float* __restrict__ W3,
                     const float* __restrict__ b3, float* __restrict__ out) {
    extern __shared__ unsigned short Tlds[];   // 128 * 520 bf16 = 133120 B
    __shared__ float logits[128];

    const int tid = threadIdx.x;
    const int j0  = blockIdx.x * 8;            // 64 j-blocks
    const int i0  = blockIdx.y * 16;           // 32 i-blocks

    if (tid < 128) logits[tid] = 0.0f;

    // ---- Stage T = bf16(tanh(Yi[i]+Yj[j])) into LDS -------------
    for (int idx = tid; idx < 128 * Hsz; idx += 256) {
        int r = idx >> 9;
        int k = idx & (Hsz - 1);
        int gi = i0 + (r >> 3);
        int gj = j0 + (r & 7);
        float v = fast_tanhf(Yi[gi * Hsz + k] + Yj[gj * Hsz + k]);
        Tlds[r * LDS_STRIDE + k] = f32_to_bf16_rne(v);
    }
    __syncthreads();

    // ---- Per-wave WMMA GEMM + fused tail ------------------------
    const int lane  = tid & 31;
    const int wv    = tid >> 5;                // 0..7
    const int mw    = wv & 3;                  // M-wave: 4 x 32 rows
    const int nw    = wv >> 2;                 // N-wave: 2 x 256 cols
    const int half  = lane >> 4;
    const int m     = lane & 15;
    const int rbase = mw * 32;
    const int nbase = nw * 256;

    float lacc0[8], lacc1[8];
    #pragma unroll
    for (int p = 0; p < 8; ++p) { lacc0[p] = 0.0f; lacc1[p] = 0.0f; }

    const unsigned short* a0Base = &Tlds[(rbase + m) * LDS_STRIDE + half * 8];
    const unsigned short* a1Base = a0Base + 16 * LDS_STRIDE;

    for (int n0 = nbase; n0 < nbase + 256; n0 += 32) {
        v8f acc00 = {0.f,0.f,0.f,0.f,0.f,0.f,0.f,0.f};
        v8f acc01 = {0.f,0.f,0.f,0.f,0.f,0.f,0.f,0.f};
        v8f acc10 = {0.f,0.f,0.f,0.f,0.f,0.f,0.f,0.f};
        v8f acc11 = {0.f,0.f,0.f,0.f,0.f,0.f,0.f,0.f};
        const unsigned short* b0B = W2T + (n0 + m) * Hsz + half * 8;
        const unsigned short* b1B = b0B + 16 * Hsz;

        #pragma unroll 2
        for (int k0 = 0; k0 < Hsz; k0 += 32) {
            FragBF a0, a1, b0, b1;
            a0.u[0] = *(const uint4*)(a0Base + k0);
            a0.u[1] = *(const uint4*)(a0Base + k0 + 16);
            a1.u[0] = *(const uint4*)(a1Base + k0);
            a1.u[1] = *(const uint4*)(a1Base + k0 + 16);
            b0.u[0] = *(const uint4*)(b0B + k0);
            b0.u[1] = *(const uint4*)(b0B + k0 + 16);
            b1.u[0] = *(const uint4*)(b1B + k0);
            b1.u[1] = *(const uint4*)(b1B + k0 + 16);

            acc00 = __builtin_amdgcn_wmma_f32_16x16x32_bf16(
                false, a0.v, false, b0.v, (short)0, acc00, false, false);
            acc01 = __builtin_amdgcn_wmma_f32_16x16x32_bf16(
                false, a0.v, false, b1.v, (short)0, acc01, false, false);
            acc10 = __builtin_amdgcn_wmma_f32_16x16x32_bf16(
                false, a1.v, false, b0.v, (short)0, acc10, false, false);
            acc11 = __builtin_amdgcn_wmma_f32_16x16x32_bf16(
                false, a1.v, false, b1.v, (short)0, acc11, false, false);
        }

        // Fused epilogue: tanh(acc + b2) * W3 -> lacc, both chunks/subtiles
        const float b2n0 = b2[n0 + m];
        const float w3n0 = W3[n0 + m];
        const float b2n1 = b2[n0 + 16 + m];
        const float w3n1 = W3[n0 + 16 + m];
        #pragma unroll
        for (int p = 0; p < 8; ++p) {
            lacc0[p] = fmaf(fast_tanhf(acc00[p] + b2n0), w3n0, lacc0[p]);
            lacc0[p] = fmaf(fast_tanhf(acc01[p] + b2n1), w3n1, lacc0[p]);
            lacc1[p] = fmaf(fast_tanhf(acc10[p] + b2n0), w3n0, lacc1[p]);
            lacc1[p] = fmaf(fast_tanhf(acc11[p] + b2n1), w3n1, lacc1[p]);
        }
    }

    // ---- Reduce over the 16 lanes of each half (cols) -----------
    #pragma unroll
    for (int off = 8; off >= 1; off >>= 1) {
        #pragma unroll
        for (int p = 0; p < 8; ++p) {
            lacc0[p] += __shfl_xor(lacc0[p], off, 16);
            lacc1[p] += __shfl_xor(lacc1[p], off, 16);
        }
    }

    // Combine the two N-waves' column-halves per row in LDS.
    if (m == 0) {
        #pragma unroll
        for (int p = 0; p < 8; ++p) {
            int r0 = rbase + half * 8 + p;
            int r1 = r0 + 16;
            atomicAdd(&logits[r0], lacc0[p]);   // ds_add_f32
            atomicAdd(&logits[r1], lacc1[p]);
        }
    }
    __syncthreads();

    if (tid < 128) {
        int r  = tid;
        int gi = i0 + (r >> 3);
        int gj = j0 + (r & 7);
        out[gi * Bsz + gj] = (gi == gj) ? -20.0f : (logits[r] + b3[0]);
    }
}

// ------------------------------------------------------------------
extern "C" void kernel_launch(void* const* d_in, const int* in_sizes, int n_in,
                              void* d_out, int out_size, void* d_ws, size_t ws_size,
                              hipStream_t stream) {
    const float* h  = (const float*)d_in[0];
    const float* W1 = (const float*)d_in[1];
    const float* b1 = (const float*)d_in[2];
    const float* W2 = (const float*)d_in[3];
    const float* b2 = (const float*)d_in[4];
    const float* W3 = (const float*)d_in[5];
    const float* b3 = (const float*)d_in[6];
    float* out = (float*)d_out;

    // Workspace layout: Yi (1MB f32) | Yj (1MB f32) | W2T (512KB bf16)
    float* Yi = (float*)d_ws;
    float* Yj = Yi + Bsz * Hsz;
    unsigned short* W2T = (unsigned short*)(Yj + Bsz * Hsz);

    prep_y_kernel<<<(2 * Bsz * Hsz) / 256, 256, 0, stream>>>(h, W1, b1, Yi, Yj);
    w2t_kernel<<<(Hsz * Hsz) / 256, 256, 0, stream>>>(W2, W2T);

    dim3 grid(Bsz / 8, Bsz / 16);              // 64 x 32 = 2048 blocks
    size_t shmem = 128 * LDS_STRIDE * sizeof(unsigned short);  // 133120 B
    pair_mlp_kernel<<<grid, 256, shmem, stream>>>(Yi, Yj, W2T, b2, W3, b3, out);
}
